// MultiHeadAttention_68307159876008
// MI455X (gfx1250) — matrix operand order repaired and checked by reading, not compiled, not run on hardware
//
#include <hip/hip_runtime.h>
#include <hip/hip_bf16.h>
#include <stdint.h>

// ---------------------------------------------------------------------------
// MultiHeadAttention with ALiBi + causal mask, CDNA5 (gfx1250) WMMA bf16 path
// B=2, S=2048, H=2048, NH=16, HD=128
// Double-buffered LDS staging: one barrier per k-step, global loads for the
// next tile overlap WMMA compute on the current tile.
// ---------------------------------------------------------------------------

#define B_   2
#define S_   2048
#define H_   2048
#define NH_  16
#define HD_  128
#define M_   (B_ * S_)          // 4096 rows for projection GEMMs
#define SCALE_ 0.08838834764831845f  // 1/sqrt(128)

typedef __attribute__((ext_vector_type(16))) __bf16 v16bf;
typedef __attribute__((ext_vector_type(8)))  float  v8f;

union FragAB {            // 16 bf16 = 8 dwords per lane (A or B operand)
    v16bf    v;
    uint32_t w[8];
    unsigned short h[16];
};

__device__ __forceinline__ unsigned short f2bf(float f) {
    union { float f; uint32_t u; } x; x.f = f;
    uint32_t r = x.u + 0x7FFFu + ((x.u >> 16) & 1u);   // round-to-nearest-even
    return (unsigned short)(r >> 16);
}

__device__ __forceinline__ v8f vzero8() {
    v8f z = {0.f, 0.f, 0.f, 0.f, 0.f, 0.f, 0.f, 0.f};
    return z;
}

__device__ __forceinline__ void load_b_frag_lds(FragAB& b, const unsigned short* p) {
    // 32 contiguous bytes: 16 bf16 K-values for this lane (B-operand layout)
    const uint4* q = (const uint4*)p;
    uint4 lo = q[0], hi = q[1];
    b.w[0] = lo.x; b.w[1] = lo.y; b.w[2] = lo.z; b.w[3] = lo.w;
    b.w[4] = hi.x; b.w[5] = hi.y; b.w[6] = hi.z; b.w[7] = hi.w;
}

__device__ __forceinline__ void load_a_frag_lds(FragAB& a, const unsigned short* row,
                                                int lane) {
    // A operand: 16x32 bf16, 16 contiguous k per lane (k = 16*(lane/16) + s)
    int kb = (lane < 16) ? 0 : 16;
    const uint32_t* r = (const uint32_t*)row;
    #pragma unroll
    for (int j = 0; j < 4; j++) a.w[j]     = r[(kb + 2 * j) >> 1];
    #pragma unroll
    for (int j = 0; j < 4; j++) a.w[4 + j] = r[(kb + 8 + 2 * j) >> 1];
}

// ---------------------------------------------------------------------------
// f32 -> bf16 bit conversion
// ---------------------------------------------------------------------------
__global__ __launch_bounds__(256) void cvt_bf16(const float* __restrict__ src,
                                                unsigned short* __restrict__ dst,
                                                int n) {
    int i = blockIdx.x * 256 + threadIdx.x;
    int stride = gridDim.x * 256;
    for (; i < n; i += stride) dst[i] = f2bf(src[i]);
}

// ---------------------------------------------------------------------------
// GEMM: C[m,n] = sum_k A[m,k] * W[n,k]    (A: MxK row-major, W: NxK row-major)
// Block tile 128(M) x 128(N) x 32(K), 256 threads = 8 waves (4 in M, 2 in N).
// Each wave: 32 rows x 64 cols = 8 x v_wmma_f32_16x16x32_bf16 per k-step.
// Double-buffered LDS, single barrier per k-step.
// MODE 0: bf16 output, permuted to [B, NH, S, HD]   (QKV projections)
// MODE 1: f32 output, row-major MxN                  (final Wo projection)
// ---------------------------------------------------------------------------
template <int MODE>
__global__ __launch_bounds__(256) void gemm_bf16_wmma(
        const unsigned short* __restrict__ A,
        const unsigned short* __restrict__ W,
        void* __restrict__ Cout,
        int M, int N, int K) {
    __shared__ unsigned short As[2][128][32];   // 16 KB
    __shared__ unsigned short Ws[2][128][32];   // 16 KB

    const int tid  = threadIdx.x;
    const int lane = tid & 31;
    const int wave = tid >> 5;
    const int wm   = wave & 3;       // 0..3  -> 32-row M sub-tile
    const int wn   = wave >> 2;      // 0..1  -> 64-col N sub-tile
    const int tileM = blockIdx.y * 128;
    const int tileN = blockIdx.x * 128;

    v8f acc[2][4];
    #pragma unroll
    for (int g = 0; g < 2; g++)
        #pragma unroll
        for (int f = 0; f < 4; f++) acc[g][f] = vzero8();

    const uint32_t* Ag = (const uint32_t*)A;   // pairs of bf16
    const uint32_t* Wg = (const uint32_t*)W;
    const int Kw = K >> 1;                     // row stride in dwords
    const int nkt = K >> 5;                    // # of 32-wide k-steps

    const int sr = tid >> 4;                   // staging row (thread's slice)
    const int sc = tid & 15;                   // staging dword col

    // ---- prologue: load k-step 0 into registers --------------------------
    uint32_t areg[8], wreg[8];
    #pragma unroll
    for (int i = 0; i < 8; i++) {
        areg[i] = Ag[(size_t)(tileM + sr + i * 16) * Kw + sc];
        wreg[i] = Wg[(size_t)(tileN + sr + i * 16) * Kw + sc];
    }

    for (int kt = 0; kt < nkt; kt++) {
        const int buf = kt & 1;
        // ---- commit staged registers to LDS ------------------------------
        uint32_t* Asw = (uint32_t*)&As[buf][0][0];
        uint32_t* Wsw = (uint32_t*)&Ws[buf][0][0];
        #pragma unroll
        for (int i = 0; i < 8; i++) {
            Asw[(sr + i * 16) * 16 + sc] = areg[i];
            Wsw[(sr + i * 16) * 16 + sc] = wreg[i];
        }
        // ---- issue global loads for next k-step (overlaps compute) -------
        if (kt + 1 < nkt) {
            int kc = ((kt + 1) << 4) + sc;     // dword column
            #pragma unroll
            for (int i = 0; i < 8; i++) {
                areg[i] = Ag[(size_t)(tileM + sr + i * 16) * Kw + kc];
                wreg[i] = Wg[(size_t)(tileN + sr + i * 16) * Kw + kc];
            }
        }
        __syncthreads();

        // ---- compute on current buffer -----------------------------------
        FragAB a[2];
        #pragma unroll
        for (int g = 0; g < 2; g++)
            load_a_frag_lds(a[g], &As[buf][wm * 32 + g * 16 + (lane & 15)][0], lane);
        #pragma unroll
        for (int f = 0; f < 4; f++) {
            FragAB b;
            int n  = wn * 64 + f * 16 + (lane & 15);
            int kb = (lane < 16) ? 0 : 16;     // 16 contiguous K per lane
            load_b_frag_lds(b, &Ws[buf][n][kb]);
            #pragma unroll
            for (int g = 0; g < 2; g++)
                acc[g][f] = __builtin_amdgcn_wmma_f32_16x16x32_bf16(
                                false, a[g].v, false, b.v, (short)0, acc[g][f], false, false);
        }
        // no trailing barrier: next store targets the other buffer; the
        // barrier at iteration kt+1 orders it against reads at kt.
    }

    // ---- store (C/D layout: lane -> col n=lane&15, rows m = 8*(lane>>4)+r)
    #pragma unroll
    for (int g = 0; g < 2; g++) {
        #pragma unroll
        for (int f = 0; f < 4; f++) {
            int n = tileN + wn * 64 + f * 16 + (lane & 15);
            #pragma unroll
            for (int r = 0; r < 8; r++) {
                int m = tileM + wm * 32 + g * 16 + 8 * (lane >> 4) + r;
                if (MODE == 0) {
                    // permute (b, s, h*HD+d) -> [B, NH, S, HD] bf16
                    int b = m >> 11, s = m & (S_ - 1);
                    int h = n >> 7, d = n & 127;
                    unsigned short* O = (unsigned short*)Cout;
                    O[(((size_t)(b * NH_ + h) * S_) + s) * HD_ + d] = f2bf(acc[g][f][r]);
                } else {
                    float* O = (float*)Cout;
                    O[(size_t)m * N + n] = acc[g][f][r];
                }
            }
        }
    }
}

// ---------------------------------------------------------------------------
// Flash attention, causal + ALiBi. One wave handles a 16-row q tile; all 8
// waves of a block share (b,h), so K and V kv-blocks are staged in LDS once
// per block (V transposed so PV B-fragments are contiguous ds_load_b128).
// Double-buffered: global loads for kv-block kb+1 overlap compute on kb.
// Q,K,V: bf16 [B,NH,S,HD].  Output O: bf16 [B,S,H].
// ---------------------------------------------------------------------------
__global__ __launch_bounds__(256) void attn_wmma(
        const unsigned short* __restrict__ Q,
        const unsigned short* __restrict__ K,
        const unsigned short* __restrict__ V,
        unsigned short* __restrict__ O) {
    __shared__ unsigned short Ks[2][32][128];    // kv block, row-major, 16 KB
    __shared__ unsigned short Vt[2][128][32];    // kv block, transposed, 16 KB
    __shared__ unsigned short pbuf[8][16][32];   // per-wave P staging,    8 KB

    const int tid  = threadIdx.x;
    const int lane = tid & 31;
    const int wave = tid >> 5;
    const int bh   = blockIdx.y;                 // b*NH + h
    const int b    = bh >> 4;
    const int h    = bh & 15;
    const int qbase = (blockIdx.x * 8 + wave) * 16;

    const unsigned short* Qh = Q + (size_t)bh * S_ * HD_;
    const uint32_t* Kg = (const uint32_t*)(K + (size_t)bh * S_ * HD_);  // 64 dw/row
    const uint32_t* Vg = (const uint32_t*)(V + (size_t)bh * S_ * HD_);

    const float slope = exp2f(-0.5f * (float)(h + 1));

    // ---- load Q A-fragments for d = 0..127 (4 k-steps), keep in registers
    FragAB aq[4];
    {
        int srow = qbase + (lane & 15);
        const unsigned short* row = Qh + (size_t)srow * HD_;
        #pragma unroll
        for (int ks = 0; ks < 4; ks++)
            load_a_frag_lds(aq[ks], row + ks * 32, lane);
    }

    v8f acc_o[8];
    #pragma unroll
    for (int dt = 0; dt < 8; dt++) acc_o[dt] = vzero8();
    float rm[8], rl[8];
    #pragma unroll
    for (int r = 0; r < 8; r++) { rm[r] = -1e30f; rl[r] = 0.f; }

    // all waves iterate the block-wide causal range (max q row in block +15)
    const int nkb = ((blockIdx.x * 128 + 112) + 16 + 31) >> 5;

    const int skv = tid >> 6;                    // staging: 4 kv rows / 64 thr
    const int scv = tid & 63;                    // dword col within row

    // ---- prologue: kv block 0 into registers -----------------------------
    uint32_t kreg[8], vreg[8];
    #pragma unroll
    for (int i = 0; i < 8; i++) {
        kreg[i] = Kg[(size_t)(skv + i * 4) * 64 + scv];
        vreg[i] = Vg[(size_t)(skv + i * 4) * 64 + scv];
    }

    for (int kb = 0; kb < nkb; kb++) {
        const int kv0 = kb * 32;
        const int buf = kb & 1;

        // ---- commit staged registers: K row-major, V transposed ----------
        {
            uint32_t* Ksw = (uint32_t*)&Ks[buf][0][0];
            #pragma unroll
            for (int i = 0; i < 8; i++) {
                int kv = skv + i * 4;
                Ksw[kv * 64 + scv] = kreg[i];
                Vt[buf][2 * scv][kv]     = (unsigned short)(vreg[i] & 0xFFFFu);
                Vt[buf][2 * scv + 1][kv] = (unsigned short)(vreg[i] >> 16);
            }
        }
        // ---- issue global loads for next kv block (overlaps compute) -----
        if (kb + 1 < nkb) {
            #pragma unroll
            for (int i = 0; i < 8; i++) {
                size_t off = (size_t)(kv0 + 32 + skv + i * 4) * 64 + scv;
                kreg[i] = Kg[off];
                vreg[i] = Vg[off];
            }
        }
        __syncthreads();

        if (kv0 <= qbase + 15) {                 // wave-uniform causal skip
            // ---- scores S = Q K^T for 16 x 32 block ----------------------
            v8f sc[2];
            sc[0] = vzero8(); sc[1] = vzero8();
            #pragma unroll
            for (int c = 0; c < 2; c++) {
                int kkl = c * 16 + (lane & 15);            // B-frag column
                #pragma unroll
                for (int ks = 0; ks < 4; ks++) {
                    int db = ks * 32 + ((lane < 16) ? 0 : 16);
                    FragAB bk;
                    load_b_frag_lds(bk, &Ks[buf][kkl][db]);
                    sc[c] = __builtin_amdgcn_wmma_f32_16x16x32_bf16(
                                false, aq[ks].v, false, bk.v, (short)0, sc[c], false, false);
                }
            }

            // ---- scale + ALiBi + causal mask, online softmax -------------
            float vals[2][8];
            float tmax[8];
            #pragma unroll
            for (int r = 0; r < 8; r++) tmax[r] = -1e30f;
            #pragma unroll
            for (int c = 0; c < 2; c++) {
                int kk = kv0 + c * 16 + (lane & 15);
                #pragma unroll
                for (int r = 0; r < 8; r++) {
                    int mrow = 8 * (lane >> 4) + r;
                    int q    = qbase + mrow;
                    float v  = sc[c][r] * SCALE_ - slope * (float)kk;
                    if (kk > q) v -= 1.0e9f;               // causal mask
                    vals[c][r] = v;
                    tmax[r] = fmaxf(tmax[r], v);
                }
            }
            #pragma unroll
            for (int r = 0; r < 8; r++) {                  // row-max, 16 lanes
                float t = tmax[r];
                #pragma unroll
                for (int s = 8; s >= 1; s >>= 1)
                    t = fmaxf(t, __shfl_xor(t, s, 16));
                tmax[r] = t;
            }
            float corr[8];
            #pragma unroll
            for (int r = 0; r < 8; r++) {
                float mnew = fmaxf(rm[r], tmax[r]);
                corr[r] = __expf(rm[r] - mnew);
                rm[r] = mnew;
            }
            float tsum[8];
            #pragma unroll
            for (int r = 0; r < 8; r++) tsum[r] = 0.f;
            #pragma unroll
            for (int c = 0; c < 2; c++) {
                #pragma unroll
                for (int r = 0; r < 8; r++) {
                    int mrow = 8 * (lane >> 4) + r;
                    float p = __expf(vals[c][r] - rm[r]);
                    tsum[r] += p;
                    pbuf[wave][mrow][c * 16 + (lane & 15)] = f2bf(p);
                }
            }
            #pragma unroll
            for (int r = 0; r < 8; r++) {                  // row-sum, 16 lanes
                float t = tsum[r];
                #pragma unroll
                for (int s = 8; s >= 1; s >>= 1)
                    t += __shfl_xor(t, s, 16);
                rl[r] = rl[r] * corr[r] + t;
            }
            #pragma unroll
            for (int dt = 0; dt < 8; dt++)
                #pragma unroll
                for (int r = 0; r < 8; r++)
                    acc_o[dt][r] *= corr[r];

            // ---- reload P as A fragment (wave-private LDS, in-order) -----
            FragAB ap;
            load_a_frag_lds(ap, &pbuf[wave][lane & 15][0], lane);

            // ---- acc_o += P (16x32) x V (32x128) via transposed LDS ------
            #pragma unroll
            for (int dt = 0; dt < 8; dt++) {
                FragAB bv;
                int d   = dt * 16 + (lane & 15);
                int kvb = (lane < 16) ? 0 : 16;
                load_b_frag_lds(bv, &Vt[buf][d][kvb]);
                acc_o[dt] = __builtin_amdgcn_wmma_f32_16x16x32_bf16(
                                false, ap.v, false, bv.v, (short)0, acc_o[dt], false, false);
            }
        }
        // no trailing barrier: next commit targets the other buffer.
    }

    // ---- normalize + store O as bf16 [B, S, H] --------------------------
    #pragma unroll
    for (int dt = 0; dt < 8; dt++) {
        #pragma unroll
        for (int r = 0; r < 8; r++) {
            int mrow = 8 * (lane >> 4) + r;
            int srow = qbase + mrow;
            float o  = acc_o[dt][r] / rl[r];
            O[((size_t)(b * S_ + srow)) * H_ + h * HD_ + dt * 16 + (lane & 15)] = f2bf(o);
        }
    }
}

// ---------------------------------------------------------------------------
// Host-side launch
// ---------------------------------------------------------------------------
extern "C" void kernel_launch(void* const* d_in, const int* in_sizes, int n_in,
                              void* d_out, int out_size, void* d_ws, size_t ws_size,
                              hipStream_t stream) {
    (void)in_sizes; (void)n_in; (void)out_size; (void)ws_size;
    const float* x  = (const float*)d_in[0];
    // d_in[1] = attention_mask, d_in[2] = alibi: recomputed analytically.
    const float* Wq = (const float*)d_in[3];
    const float* Wk = (const float*)d_in[4];
    const float* Wv = (const float*)d_in[5];
    const float* Wo = (const float*)d_in[6];

    char* ws = (char*)d_ws;
    const size_t MB = 1024ull * 1024ull;
    unsigned short* xb  = (unsigned short*)(ws + 0 * MB);    // 16 MiB  (M x H)
    unsigned short* wqb = (unsigned short*)(ws + 16 * MB);   //  8 MiB
    unsigned short* wkb = (unsigned short*)(ws + 24 * MB);
    unsigned short* wvb = (unsigned short*)(ws + 32 * MB);
    unsigned short* wob = (unsigned short*)(ws + 40 * MB);
    unsigned short* Qb  = (unsigned short*)(ws + 48 * MB);   // 16 MiB  [B,NH,S,HD]
    unsigned short* Kb  = (unsigned short*)(ws + 64 * MB);
    unsigned short* Vb  = (unsigned short*)(ws + 80 * MB);
    unsigned short* Ob  = (unsigned short*)(ws + 96 * MB);   // 16 MiB  [B,S,H]

    const int nx = M_ * H_;        // 8,388,608
    const int nw = H_ * H_;        // 4,194,304
    cvt_bf16<<<1024, 256, 0, stream>>>(x,  xb,  nx);
    cvt_bf16<<<1024, 256, 0, stream>>>(Wq, wqb, nw);
    cvt_bf16<<<1024, 256, 0, stream>>>(Wk, wkb, nw);
    cvt_bf16<<<1024, 256, 0, stream>>>(Wv, wvb, nw);
    cvt_bf16<<<1024, 256, 0, stream>>>(Wo, wob, nw);

    dim3 gGemm(H_ / 128, M_ / 128);                // 16 x 32 blocks
    gemm_bf16_wmma<0><<<gGemm, 256, 0, stream>>>(xb, wqb, Qb, M_, H_, H_);
    gemm_bf16_wmma<0><<<gGemm, 256, 0, stream>>>(xb, wkb, Kb, M_, H_, H_);
    gemm_bf16_wmma<0><<<gGemm, 256, 0, stream>>>(xb, wvb, Vb, M_, H_, H_);

    dim3 gAttn(S_ / 128, B_ * NH_);                // 16 x 32 blocks, 8 q-tiles/block
    attn_wmma<<<gAttn, 256, 0, stream>>>(Qb, Kb, Vb, Ob);

    gemm_bf16_wmma<1><<<gGemm, 256, 0, stream>>>(Ob, wob, (float*)d_out, M_, H_, H_);
}
